// BagModel_57698590654615
// MI455X (gfx1250) — matrix operand order repaired and checked by reading, not compiled, not run on hardware
//
#include <hip/hip_runtime.h>
#include <hip/hip_bf16.h>

typedef __attribute__((ext_vector_type(16))) __bf16 v16bf;
typedef __attribute__((ext_vector_type(8)))  float  v8f;
typedef __attribute__((ext_vector_type(4)))  float  v4f;

#define D_IN   512
#define D_H    1000
#define D_HP   1008          // padded to 63*16
#define NCHUNK 63            // N-dim 16-wide chunks
#define KCHUNK 16            // K-dim 32-wide chunks (512/32)

// workspace layout (bytes)
#define OFF_B      0u                      // packed bf16 B fragments: 1008*1024 = 1032192
#define OFF_BP     (1u << 20)              // 1008 floats
#define OFF_WG     ((1u << 20) + 4096u)    // 1008 floats
#define OFF_LOGITS ((1u << 20) + 16384u)   // N floats
#define OFF_ARG    (OFF_LOGITS + 400000u)  // u64 argmax cell

__device__ __forceinline__ unsigned pack2bf(float a, float b) {
    unsigned short lo = __builtin_bit_cast(unsigned short, (__bf16)a);
    unsigned short hi = __builtin_bit_cast(unsigned short, (__bf16)b);
    return (unsigned)lo | ((unsigned)hi << 16);
}

// ---------------- pack Wp -> bf16 WMMA-B fragments; pad bp/Wg; reset argmax ----
__global__ void pack_kernel(const float* __restrict__ Wp,
                            const float* __restrict__ bp,
                            const float* __restrict__ Wg,
                            char* __restrict__ ws) {
    int b = blockIdx.x, t = threadIdx.x;
    if (b < NCHUNK * KCHUNK) {
        if (t >= 32) return;
        int c  = b >> 4;                       // N-chunk
        int kk = b & 15;                       // K-chunk
        int n  = c * 16 + (t & 15);            // column
        int kb = kk * 32 + ((t >> 4) << 4);    // lanes 0-15: K 0..15; 16-31: K 16..31
        unsigned* dst = (unsigned*)(ws + ((unsigned)b << 10) + (unsigned)t * 32u);
#pragma unroll
        for (int v = 0; v < 8; ++v) {
            int k0 = kb + 2 * v;
            float f0 = (n < D_H) ? Wp[k0 * D_H + n]       : 0.f;
            float f1 = (n < D_H) ? Wp[(k0 + 1) * D_H + n] : 0.f;
            dst[v] = pack2bf(f0, f1);
        }
    } else {
        float* obp = (float*)(ws + OFF_BP);
        float* owg = (float*)(ws + OFF_WG);
        for (int j = t; j < D_HP; j += 64) {
            obp[j] = (j < D_H) ? bp[j] : 0.f;
            owg[j] = (j < D_H) ? Wg[j] : 0.f;
        }
        if (t == 0) *(unsigned long long*)(ws + OFF_ARG) = 0ull;
    }
}

// ---------------- fused GEMM + relu + dot(Wg) -> logits --------------------------
__global__ __launch_bounds__(256) void logits_kernel(const float* __restrict__ x,
                                                     const char*  __restrict__ wsB,
                                                     const float* __restrict__ wsBP,
                                                     const float* __restrict__ wsWG,
                                                     float* __restrict__ logits,
                                                     int n) {
    __shared__ __align__(16) unsigned char sA[128 * 1024];  // 128 rows x 512 K bf16, fragment swizzle
    __shared__ float sBP[D_HP];
    __shared__ float sWG[D_HP];

    const int tid   = threadIdx.x;
    const int rows0 = blockIdx.x * 128;

    // Phase A: stage x tile fp32 -> bf16 LDS in 16-bit A-matrix fragment layout
    for (int it = 0; it < 64; ++it) {
        int lid = (it << 8) + tid;
        int c4  = lid & 127;       // float4 column
        int m   = lid >> 7;        // 0..127 (threads 0-127 cover a full row -> coalesced)
        int row = rows0 + m;
        v4f v = {0.f, 0.f, 0.f, 0.f};
        if (row < n) v = *(const v4f*)(x + (size_t)row * D_IN + (c4 << 2));
        int k  = c4 << 2;
        int mt = m >> 4, mm = m & 15, kk = k >> 5, kq = k & 31;
        int lane_ = mm + ((kq & 16) ? 16 : 0);
        int vg    = (kq & 15) >> 1;
        int off   = ((mt * 16 + kk) << 10) + lane_ * 32 + vg * 4;
        uint2 pk;
        pk.x = pack2bf(v.x, v.y);
        pk.y = pack2bf(v.z, v.w);
        *(uint2*)(sA + off) = pk;
    }
    for (int j = tid; j < D_HP; j += 256) { sBP[j] = wsBP[j]; sWG[j] = wsWG[j]; }
    __syncthreads();

    // Phase B: each wave owns a 16-row M tile; cache its 16 A fragments in VGPRs
    const int wv = tid >> 5, lane = tid & 31;
    v16bf a[KCHUNK];
#pragma unroll
    for (int kk = 0; kk < KCHUNK; ++kk)
        a[kk] = *(const v16bf*)(sA + ((wv * 16 + kk) << 10) + lane * 32);

    float acc[8] = {0.f, 0.f, 0.f, 0.f, 0.f, 0.f, 0.f, 0.f};
    for (int nc = 0; nc < NCHUNK; ++nc) {
        const char* bbase = wsB + ((unsigned)(nc * 16) << 10) + (unsigned)lane * 32u;
        if (nc + 1 < NCHUNK) __builtin_prefetch(bbase + (16 << 10), 0, 0);
        v8f c = {};
#pragma unroll
        for (int kk = 0; kk < KCHUNK; ++kk) {
            v16bf bfrag = *(const v16bf*)(bbase + (kk << 10));
            c = __builtin_amdgcn_wmma_f32_16x16x32_bf16(false, a[kk], false, bfrag,
                                                        (short)0, c, false, false);
        }
        // epilogue: relu(c + bp) * Wg, accumulate per M-row (C layout: VGPR r, lane l:
        // M = r + 8*(l>=16), Ncol = l%16)
        int   ncol = nc * 16 + (lane & 15);
        float bpv  = sBP[ncol];
        float wgv  = sWG[ncol];
#pragma unroll
        for (int r = 0; r < 8; ++r) {
            float h = c[r] + bpv;
            h = h > 0.f ? h : 0.f;
            acc[r] += h * wgv;
        }
    }
    // reduce across the 16 lanes of each half (xor masks 1,2,4,8 stay within a half)
#pragma unroll
    for (int r = 0; r < 8; ++r) {
        float s = acc[r];
        s += __shfl_xor(s, 1);
        s += __shfl_xor(s, 2);
        s += __shfl_xor(s, 4);
        s += __shfl_xor(s, 8);
        acc[r] = s;
    }
    if ((lane & 15) == 0) {
        int mbase = rows0 + wv * 16 + ((lane >> 4) << 3);
#pragma unroll
        for (int r = 0; r < 8; ++r) {
            int row = mbase + r;
            if (row < n) logits[row] = acc[r];
        }
    }
}

// ---------------- argmax via monotone u64 keys + atomicMax -----------------------
__global__ void argmax_kernel(const float* __restrict__ logits,
                              unsigned long long* __restrict__ cell, int n) {
    __shared__ unsigned long long red[256];
    int tid = threadIdx.x;
    unsigned long long best = 0ull;
    for (int i = blockIdx.x * 256 + tid; i < n; i += gridDim.x * 256) {
        unsigned b = __float_as_uint(logits[i]);
        b = (b & 0x80000000u) ? ~b : (b | 0x80000000u);   // monotone float->uint
        unsigned long long key = ((unsigned long long)b << 32) |
                                 (unsigned long long)(~(unsigned)i); // ties -> lowest i
        best = key > best ? key : best;
    }
    red[tid] = best;
    __syncthreads();
    for (int s = 128; s > 0; s >>= 1) {
        if (tid < s && red[tid + s] > red[tid]) red[tid] = red[tid + s];
        __syncthreads();
    }
    if (tid == 0) atomicMax(cell, red[0]);
}

// ---------------- recompute chosen row in fp32; emit 256 outputs -----------------
__global__ void finalize_kernel(const float* __restrict__ x,
                                const float* __restrict__ Wp,
                                const float* __restrict__ bp,
                                const float* __restrict__ Wa,
                                const float* __restrict__ ba,
                                const unsigned long long* __restrict__ cell,
                                float* __restrict__ out, int out_n) {
    __shared__ float xr[D_IN];
    __shared__ float red[256];
    const int tid = threadIdx.x;
    const int row = (int)(~(unsigned)(*cell));   // decode index
    for (int k = tid; k < D_IN; k += 256) xr[k] = x[(size_t)row * D_IN + k];
    __syncthreads();
    float part = 0.f;
    for (int j = tid; j < D_H; j += 256) {
        float d = bp[j];
        for (int k = 0; k < D_IN; ++k) d += xr[k] * Wp[k * D_H + j];
        d = d > 0.f ? d : 0.f;
        part += d * Wa[j];
    }
    red[tid] = part;
    __syncthreads();
    for (int s = 128; s > 0; s >>= 1) {
        if (tid < s) red[tid] += red[tid + s];
        __syncthreads();
    }
    float b0 = ba[0];
    if (tid < out_n) out[tid] = (tid == 0) ? (red[0] + b0) : b0;
}

extern "C" void kernel_launch(void* const* d_in, const int* in_sizes, int n_in,
                              void* d_out, int out_size, void* d_ws, size_t ws_size,
                              hipStream_t stream) {
    const float* x  = (const float*)d_in[0];
    // d_in[1] = ids (unused: argmax of a global softmax ignores bag ids)
    const float* Wp = (const float*)d_in[2];
    const float* bp = (const float*)d_in[3];
    const float* Wg = (const float*)d_in[4];
    // d_in[5] = bg (constant shift, cannot change argmax; not in output)
    const float* Wa = (const float*)d_in[6];
    const float* ba = (const float*)d_in[7];
    char* ws = (char*)d_ws;

    const int n = in_sizes[0] / D_IN;
    const int gridM = (n + 127) / 128;

    pack_kernel<<<NCHUNK * KCHUNK + 1, 64, 0, stream>>>(Wp, bp, Wg, ws);
    logits_kernel<<<gridM, 256, 0, stream>>>(x, ws,
                                             (const float*)(ws + OFF_BP),
                                             (const float*)(ws + OFF_WG),
                                             (float*)(ws + OFF_LOGITS), n);
    argmax_kernel<<<256, 256, 0, stream>>>((const float*)(ws + OFF_LOGITS),
                                           (unsigned long long*)(ws + OFF_ARG), n);
    finalize_kernel<<<1, 256, 0, stream>>>(x, Wp, bp, Wa, ba,
                                           (const unsigned long long*)(ws + OFF_ARG),
                                           (float*)d_out, out_size);
}
